// GraphMoECaseSize_10101763080595
// MI455X (gfx1250) — matrix operand order repaired and checked by reading, not compiled, not run on hardware
//
#include <hip/hip_runtime.h>

// ---------------------------------------------------------------------------
// Problem constants (match reference)
// ---------------------------------------------------------------------------
#define N_NODES 30000
#define E_EDGES 960000
#define G_GRAPHS 16
#define DH 256
#define N_EXP 8
#define L_LAYERS 3

typedef __attribute__((ext_vector_type(16))) __bf16 bf16x16;
typedef __attribute__((ext_vector_type(8)))  __bf16 bf16x8;
typedef __attribute__((ext_vector_type(8)))  float  floatx8;
typedef unsigned int uint32x4 __attribute__((ext_vector_type(4)));
typedef int          int32x8  __attribute__((ext_vector_type(8)));
typedef int          int32x4  __attribute__((ext_vector_type(4)));

// ---------------------------------------------------------------------------
// bf16 <-> f32 helpers (bit-exact RNE, no reliance on native cast legality)
// ---------------------------------------------------------------------------
__device__ __forceinline__ __bf16 f2bf(float f) {
  unsigned int u = __builtin_bit_cast(unsigned int, f);
  u += 0x7fffu + ((u >> 16) & 1u);            // round-to-nearest-even
  unsigned short h = (unsigned short)(u >> 16);
  return __builtin_bit_cast(__bf16, h);
}
__device__ __forceinline__ float bf2f(__bf16 b) {
  unsigned short h = __builtin_bit_cast(unsigned short, b);
  unsigned int u = ((unsigned int)h) << 16;
  return __builtin_bit_cast(float, u);
}
__device__ __forceinline__ void atomAddF(float* p, float v) {
  __hip_atomic_fetch_add(p, v, __ATOMIC_RELAXED, __HIP_MEMORY_SCOPE_AGENT);
}

// ---------------------------------------------------------------------------
// WMMA fragment loaders (contiguous-K bf16 layouts, not the ISA table order; wave32)
// A 16x32 bf16: lane L -> row L%16; K = (L/16)*16 .. +15 contiguous
// B 32x16 bf16 (from W^T row-major [out][K]): lane L -> col L%16,
//   K = (L/16)*16 .. +15 contiguous
// ---------------------------------------------------------------------------
__device__ __forceinline__ bf16x16 load_a_frag(const __bf16* __restrict__ rowp,
                                               int k0, int lane) {
  const int kk = k0 + ((lane >> 4) << 4);
  bf16x8 lo = *(const bf16x8*)(rowp + kk);
  bf16x8 hi = *(const bf16x8*)(rowp + kk + 8);
  bf16x16 f;
#pragma unroll
  for (int i = 0; i < 8; ++i) { f[i] = lo[i]; f[i + 8] = hi[i]; }
  return f;
}

// B fragment from a per-lane LDS base pointer (immediate-offset friendly)
__device__ __forceinline__ bf16x16 ldb(const __bf16* p) {
  bf16x8 lo = *(const bf16x8*)p;
  bf16x8 hi = *(const bf16x8*)(p + 8);
  bf16x16 f;
#pragma unroll
  for (int i = 0; i < 8; ++i) { f[i] = lo[i]; f[i + 8] = hi[i]; }
  return f;
}

__device__ __forceinline__ floatx8 wmma_bf16(bf16x16 a, bf16x16 b, floatx8 c) {
  return __builtin_amdgcn_wmma_f32_16x16x32_bf16(false, a, false, b,
                                                 (short)0, c, false, false);
}

// ---------------------------------------------------------------------------
// Tensor Data Mover: stage one 256x256 bf16 panel (128 KB, contiguous) into
// LDS. D# built per ISA cdna5 ch.8: group0 = {flags, lds_addr, global_addr,
// type=2}, group1 = {data_size=2B, tensor/tile dims 256x256, stride 256}.
// Groups 2/3 zero (2-D tensor). Tracked by TENSORcnt.
// This toolchain's builtin is the 6-arg form:
//   (uint32x4 g0, int32x8 g1, int32x4 g2, int32x4 g3, int32x8 pad, i32 cpol)
// ---------------------------------------------------------------------------
__device__ __forceinline__ void tdm_load_panel_256x256_bf16(
    const void* gsrc, unsigned lds_byte_off) {
  const unsigned long long ga = (unsigned long long)gsrc;
  uint32x4 g0;
  g0[0] = 1u;                                       // count=1, user descriptor
  g0[1] = lds_byte_off;                             // lds_addr
  g0[2] = (unsigned)(ga & 0xFFFFFFFFu);             // global_addr[31:0]
  g0[3] = (unsigned)((ga >> 32) & 0x01FFFFFFu)      // global_addr[56:32]
        | (2u << 30);                               // type = 2 ("image")
  int32x8 g1;
  g1[0] = 1 << 16;                                  // data_size=1 -> 2 bytes
  g1[1] = (256 & 0xFFFF) << 16;                     // tensor_dim0 = 256
  g1[2] = (256 & 0xFFFF) << 16;                     // tensor_dim1 = 256
  g1[3] = (256 & 0xFFFF) << 16;                     // tile_dim0 = 256
  g1[4] = 256;                                      // tile_dim1 = 256
  g1[5] = 256;                                      // tensor_dim0_stride = 256
  g1[6] = 0;
  g1[7] = 0;
  int32x4 z4 = {0, 0, 0, 0};
  int32x8 z8 = {0, 0, 0, 0, 0, 0, 0, 0};
  __builtin_amdgcn_tensor_load_to_lds(g0, g1, z4, z4, z8, 0);
}

// ---------------------------------------------------------------------------
// Encoder: h = relu(x @ enc_W + enc_b), stored bf16. One block per node.
// ---------------------------------------------------------------------------
__global__ void encoder_kernel(const float* __restrict__ x,
                               const float* __restrict__ W,
                               const float* __restrict__ b,
                               __bf16* __restrict__ h) {
  const int node = blockIdx.x;
  const int j = threadIdx.x;
  __shared__ float xs[6];
  if (j < 6) xs[j] = x[node * 6 + j];
  __syncthreads();
  float acc = b[j];
#pragma unroll
  for (int k = 0; k < 6; ++k) acc = fmaf(xs[k], W[k * DH + j], acc);
  h[(size_t)node * DH + j] = f2bf(fmaxf(acc, 0.f));
}

// ---------------------------------------------------------------------------
// Graph size stats: gcnt[0..15] = #nodes per graph, gcnt[16..31] = #edges.
// ---------------------------------------------------------------------------
__global__ void stats_kernel(const long long* __restrict__ src,
                             const long long* __restrict__ batch,
                             float* __restrict__ gcnt) {
  __shared__ float sc[32];
  const int tid = threadIdx.x;
  if (tid < 32) sc[tid] = 0.f;
  __syncthreads();
  const int idx = blockIdx.x * blockDim.x + tid;
  const int stride = gridDim.x * blockDim.x;
  for (int i = idx; i < E_EDGES; i += stride)
    atomicAdd(&sc[16 + (int)batch[(int)src[i]]], 1.f);
  for (int i = idx; i < N_NODES; i += stride)
    atomicAdd(&sc[(int)batch[i]], 1.f);
  __syncthreads();
  if (tid < 32) atomAddF(&gcnt[tid], sc[tid]);
}

__global__ void glog_kernel(const float* __restrict__ gcnt,
                            float* __restrict__ glog) {
  const int i = threadIdx.x;
  if (i < G_GRAPHS) {
    glog[2 * i]     = logf(fmaxf(gcnt[i], 1.f));
    glog[2 * i + 1] = logf(fmaxf(gcnt[16 + i], 1.f));
  }
}

// ---------------------------------------------------------------------------
// Weight transpose: f32 [mat][in=256][out=256] -> bf16 [mat][out=256][in=256]
// ---------------------------------------------------------------------------
__global__ void transpose_w_kernel(const float* __restrict__ src,
                                   __bf16* __restrict__ dst, int nmat) {
  const int idx = blockIdx.x * blockDim.x + threadIdx.x;
  if (idx >= nmat * 65536) return;
  const int m = idx >> 16;
  const int r = idx & 65535;
  const int o = r >> 8;     // out index
  const int i = r & 255;    // in index
  dst[idx] = f2bf(src[(m << 16) + (i << 8) + o]);
}

// ---------------------------------------------------------------------------
// Router GEMM: r_h = relu([h | size_feat] @ W1 + b1) stored bf16.
// K=256 WMMA with W1[0:256]^T in LDS (manual staging); bias in acc init; rank-2 tail in
// the f32 epilogue. Two LDS base pointers keep all ds offsets < 64 KB; k-loop
// fully unrolled so every ds/global offset is an immediate.
// ---------------------------------------------------------------------------
__global__ __launch_bounds__(256) void router_gemm_kernel(
    const __bf16* __restrict__ A, const __bf16* __restrict__ Bt,
    const float* __restrict__ bias, const float* __restrict__ rW1,
    const long long* __restrict__ batch, const float* __restrict__ glog,
    __bf16* __restrict__ Out, int nrows) {
  extern __shared__ __bf16 sB[];                 // 65536 elems = 128 KB
  const int tid = threadIdx.x;
  {
    const uint4* s = (const uint4*)Bt;
    uint4* d = (uint4*)sB;
    for (int i = tid; i < 8192; i += 256) d[i] = s[i];
  }
  __syncthreads();

  const int lane = tid & 31;
  const int wave = tid >> 5;
  const int row0 = blockIdx.x * 128 + wave * 16;
  int arow = row0 + (lane & 15);
  if (arow >= nrows) arow = nrows - 1;           // clamp: EXEC stays all-1s
  const __bf16* ap = A + (size_t)arow * DH;

  const int lanebase = ((lane & 15) << 8) + ((lane >> 4) << 4);
  const __bf16* pB0 = sB + lanebase;             // tiles 0..7
  const __bf16* pB1 = sB + 32768 + lanebase;     // tiles 8..15

  floatx8 acc[16];
#pragma unroll
  for (int t = 0; t < 16; ++t) {
    const float bv = bias[t * 16 + (lane & 15)];
#pragma unroll
    for (int v = 0; v < 8; ++v) acc[t][v] = bv;
  }

#pragma unroll
  for (int k0 = 0; k0 < DH; k0 += 32) {
    bf16x16 a = load_a_frag(ap, k0, lane);
#pragma unroll
    for (int t = 0; t < 8; ++t)
      acc[t] = wmma_bf16(a, ldb(pB0 + t * 4096 + k0), acc[t]);
#pragma unroll
    for (int t = 0; t < 8; ++t)
      acc[t + 8] = wmma_bf16(a, ldb(pB1 + t * 4096 + k0), acc[t + 8]);
  }

  const int colbase = lane & 15;
#pragma unroll
  for (int v = 0; v < 8; ++v) {
    const int r = row0 + ((lane >> 4) << 3) + v;
    if (r >= nrows) break;
    const int g = (int)batch[r];
    const float sf0 = glog[2 * g], sf1 = glog[2 * g + 1];
    __bf16* op = Out + (size_t)r * DH;
#pragma unroll
    for (int t = 0; t < 16; ++t) {
      const int col = t * 16 + colbase;
      float val = acc[t][v] + sf0 * rW1[256 * 256 + col]
                            + sf1 * rW1[257 * 256 + col];
      op[col] = f2bf(fmaxf(val, 0.f));
    }
  }
}

// ---------------------------------------------------------------------------
// Router head: logits = r_h @ W2 + b2 ; top-2 of softmax, renormalized.
// Softmax denominator cancels in the top-2 ratio.
// ---------------------------------------------------------------------------
__global__ void router_topk_kernel(const __bf16* __restrict__ rh,
                                   const float* __restrict__ W2,
                                   const float* __restrict__ b2,
                                   float* __restrict__ sparse, int n) {
  const int node = blockIdx.x * blockDim.x + threadIdx.x;
  if (node >= n) return;
  float lg[N_EXP];
#pragma unroll
  for (int j = 0; j < N_EXP; ++j) lg[j] = b2[j];
  const __bf16* hr = rh + (size_t)node * DH;
  for (int k = 0; k < DH; ++k) {
    const float hv = bf2f(hr[k]);
    const float* w = W2 + k * N_EXP;
#pragma unroll
    for (int j = 0; j < N_EXP; ++j) lg[j] = fmaf(hv, w[j], lg[j]);
  }
  int i1 = 0;
#pragma unroll
  for (int j = 1; j < N_EXP; ++j) if (lg[j] > lg[i1]) i1 = j;
  int i2 = (i1 == 0) ? 1 : 0;
#pragma unroll
  for (int j = 0; j < N_EXP; ++j)
    if (j != i1 && lg[j] > lg[i2]) i2 = j;
  const float e2 = expf(lg[i2] - lg[i1]);
  const float inv = 1.f / (1.f + e2);
  float outv[N_EXP];
#pragma unroll
  for (int j = 0; j < N_EXP; ++j) outv[j] = 0.f;
  outv[i1] = inv;
  outv[i2] = e2 * inv;
  float* op = sparse + (size_t)node * N_EXP;
#pragma unroll
  for (int j = 0; j < N_EXP; ++j) op[j] = outv[j];
}

// ---------------------------------------------------------------------------
// Edge aggregation: out[dst] += feat[src] (f32 atomics, bf16 source).
// ---------------------------------------------------------------------------
__global__ void agg_kernel(const __bf16* __restrict__ feat,
                           const long long* __restrict__ src,
                           const long long* __restrict__ dst,
                           float* __restrict__ out) {
  const int idx = blockIdx.x * blockDim.x + threadIdx.x;
  if (idx >= E_EDGES * 32) return;
  const int e = idx >> 5, c = idx & 31;
  const int s = (int)src[e], d = (int)dst[e];
  bf16x8 v = *(const bf16x8*)(feat + (size_t)s * DH + c * 8);
  float* o = out + (size_t)d * DH + c * 8;
#pragma unroll
  for (int j = 0; j < 8; ++j) {
    __bf16 t = v[j];
    atomAddF(o + j, bf2f(t));
  }
}

__global__ void f32_to_bf16_kernel(const float* __restrict__ src,
                                   __bf16* __restrict__ dst, int n8) {
  const int idx = blockIdx.x * blockDim.x + threadIdx.x;
  if (idx >= n8) return;
  const float4 a = ((const float4*)src)[idx * 2];
  const float4 b = ((const float4*)src)[idx * 2 + 1];
  bf16x8 o;
  o[0] = f2bf(a.x); o[1] = f2bf(a.y); o[2] = f2bf(a.z); o[3] = f2bf(a.w);
  o[4] = f2bf(b.x); o[5] = f2bf(b.y); o[6] = f2bf(b.z); o[7] = f2bf(b.w);
  ((bf16x8*)dst)[idx] = o;
}

// ---------------------------------------------------------------------------
// Expert GEMM: Out = [relu]( A1 @ B1 + A2 @ B2 + bias ), all [N,256]x[256,256].
// Both 128 KB bf16 weight panels staged to LDS by the Tensor Data Mover
// (one tensor_load_to_lds per panel, TENSORcnt-synchronized), 256 KB of the
// 320 KB/WGP LDS. 8 waves x 16-row stripes; 16 f32 accumulator tiles per
// wave; fully unrolled K so every ds/global offset is an immediate.
// ---------------------------------------------------------------------------
__global__ __launch_bounds__(256) void expert_gemm_kernel(
    const __bf16* __restrict__ A1, const __bf16* __restrict__ A2,
    const __bf16* __restrict__ B1t, const __bf16* __restrict__ B2t,
    const float* __restrict__ bias, __bf16* __restrict__ Out,
    int nrows, int do_relu) {
  extern __shared__ __bf16 sB[];                 // 2 * 65536 elems = 256 KB
  const int tid = threadIdx.x;
  if (tid == 0) {
    const unsigned lds0 = (unsigned)(unsigned long long)(void*)sB;
    tdm_load_panel_256x256_bf16(B1t, lds0);
    tdm_load_panel_256x256_bf16(B2t, lds0 + 131072u);
    __builtin_amdgcn_s_wait_tensorcnt(0);
  }
  __syncthreads();

  const int lane = tid & 31;
  const int wave = tid >> 5;
  const int row0 = blockIdx.x * 128 + wave * 16;
  int arow = row0 + (lane & 15);
  if (arow >= nrows) arow = nrows - 1;
  const __bf16* a1p = A1 + (size_t)arow * DH;
  const __bf16* a2p = A2 + (size_t)arow * DH;

  const int lanebase = ((lane & 15) << 8) + ((lane >> 4) << 4);
  const __bf16* pB0 = sB + lanebase;             // matrix 1, tiles 0..7
  const __bf16* pB1 = sB + 32768 + lanebase;     // matrix 1, tiles 8..15
  const __bf16* pC0 = sB + 65536 + lanebase;     // matrix 2, tiles 0..7
  const __bf16* pC1 = sB + 98304 + lanebase;     // matrix 2, tiles 8..15

  floatx8 acc[16];
#pragma unroll
  for (int t = 0; t < 16; ++t) {
    const float bv = bias[t * 16 + (lane & 15)];
#pragma unroll
    for (int v = 0; v < 8; ++v) acc[t][v] = bv;
  }

#pragma unroll
  for (int k0 = 0; k0 < DH; k0 += 32) {
    bf16x16 a1 = load_a_frag(a1p, k0, lane);
#pragma unroll
    for (int t = 0; t < 8; ++t)
      acc[t] = wmma_bf16(a1, ldb(pB0 + t * 4096 + k0), acc[t]);
#pragma unroll
    for (int t = 0; t < 8; ++t)
      acc[t + 8] = wmma_bf16(a1, ldb(pB1 + t * 4096 + k0), acc[t + 8]);
    bf16x16 a2 = load_a_frag(a2p, k0, lane);
#pragma unroll
    for (int t = 0; t < 8; ++t)
      acc[t] = wmma_bf16(a2, ldb(pC0 + t * 4096 + k0), acc[t]);
#pragma unroll
    for (int t = 0; t < 8; ++t)
      acc[t + 8] = wmma_bf16(a2, ldb(pC1 + t * 4096 + k0), acc[t + 8]);
  }

  const int colbase = lane & 15;
#pragma unroll
  for (int v = 0; v < 8; ++v) {
    const int r = row0 + ((lane >> 4) << 3) + v;
    if (r >= nrows) break;
    __bf16* op = Out + (size_t)r * DH;
#pragma unroll
    for (int t = 0; t < 16; ++t) {
      float val = acc[t][v];
      if (do_relu) val = fmaxf(val, 0.f);
      op[t * 16 + colbase] = f2bf(val);
    }
  }
}

// ---------------------------------------------------------------------------
// Combine: out[n,:] += sparse[n,e] * H[n,:]  (skip rows with zero weight)
// ---------------------------------------------------------------------------
__global__ void combine_kernel(const __bf16* __restrict__ H,
                               const float* __restrict__ sparse, int expert,
                               float* __restrict__ out, int n) {
  const int idx = blockIdx.x * blockDim.x + threadIdx.x;
  if (idx >= n * 32) return;
  const int node = idx >> 5, c = idx & 31;
  const float w = sparse[(size_t)node * N_EXP + expert];
  if (w == 0.f) return;
  bf16x8 v = *(const bf16x8*)(H + (size_t)node * DH + c * 8);
  float4* op = (float4*)(out + (size_t)node * DH + c * 8);
  float4 o0 = op[0];
  float4 o1 = op[1];
  { __bf16 t;
    t = v[0]; o0.x += w * bf2f(t);
    t = v[1]; o0.y += w * bf2f(t);
    t = v[2]; o0.z += w * bf2f(t);
    t = v[3]; o0.w += w * bf2f(t);
    t = v[4]; o1.x += w * bf2f(t);
    t = v[5]; o1.y += w * bf2f(t);
    t = v[6]; o1.z += w * bf2f(t);
    t = v[7]; o1.w += w * bf2f(t);
  }
  op[0] = o0;
  op[1] = o1;
}

// ---------------------------------------------------------------------------
// Host orchestration
// ---------------------------------------------------------------------------
extern "C" void kernel_launch(void* const* d_in, const int* in_sizes, int n_in,
                              void* d_out, int out_size, void* d_ws,
                              size_t ws_size, hipStream_t stream) {
  (void)in_sizes; (void)n_in; (void)out_size; (void)ws_size;

  const float*     x      = (const float*)d_in[0];
  const long long* eidx   = (const long long*)d_in[1];
  const long long* batch  = (const long long*)d_in[2];
  const float*     enc_W  = (const float*)d_in[3];
  const float*     enc_b  = (const float*)d_in[4];
  const float*     r_W1   = (const float*)d_in[5];
  const float*     r_b1   = (const float*)d_in[6];
  const float*     r_W2   = (const float*)d_in[7];
  const float*     r_b2   = (const float*)d_in[8];
  const float*     Wrel   = (const float*)d_in[9];
  const float*     brel   = (const float*)d_in[10];
  const float*     Wroot  = (const float*)d_in[11];
  float*           out    = (float*)d_out;

  const long long* src = eidx;
  const long long* dst = eidx + E_EDGES;

  // Workspace carve-up (256B aligned slices)
  char* w = (char*)d_ws;
  auto alloc = [&](size_t bytes) -> void* {
    void* p = (void*)w;
    w += (bytes + 255) & ~(size_t)255;
    return p;
  };
  const size_t NF  = (size_t)N_NODES * DH;
  __bf16* h_bf    = (__bf16*)alloc(NF * 2);
  __bf16* agg0_bf = (__bf16*)alloc(NF * 2);
  __bf16* A_bf    = (__bf16*)alloc(NF * 2);
  __bf16* Hc      = (__bf16*)alloc(NF * 2);
  __bf16* Hn      = (__bf16*)alloc(NF * 2);
  __bf16* rh_bf   = (__bf16*)alloc(NF * 2);
  float*  A_f32   = (float*)alloc(NF * 4);
  float*  sparse  = (float*)alloc((size_t)N_NODES * N_EXP * 4);
  float*  gcnt    = (float*)alloc(32 * 4);
  float*  glog    = (float*)alloc(32 * 4);
  __bf16* wrel_t  = (__bf16*)alloc((size_t)L_LAYERS * N_EXP * 65536 * 2);
  __bf16* wroot_t = (__bf16*)alloc((size_t)L_LAYERS * N_EXP * 65536 * 2);
  __bf16* w1_t    = (__bf16*)alloc((size_t)65536 * 2);

  const int NMAT = L_LAYERS * N_EXP;              // 24 matrices of 256x256
  const int gemm_blocks = (N_NODES + 127) / 128;  // 235
  const int agg_blocks  = (E_EDGES * 32 + 255) / 256;
  const int cvt_blocks  = ((N_NODES * 32) + 255) / 256;

  // --- One-time prep -------------------------------------------------------
  (void)hipMemsetAsync(gcnt, 0, 32 * 4, stream);
  (void)hipMemsetAsync(out, 0, NF * 4, stream);

  encoder_kernel<<<N_NODES, 256, 0, stream>>>(x, enc_W, enc_b, h_bf);
  stats_kernel<<<1024, 256, 0, stream>>>(src, batch, gcnt);
  glog_kernel<<<1, 32, 0, stream>>>(gcnt, glog);

  transpose_w_kernel<<<(NMAT * 65536) / 256, 256, 0, stream>>>(Wrel, wrel_t, NMAT);
  transpose_w_kernel<<<(NMAT * 65536) / 256, 256, 0, stream>>>(Wroot, wroot_t, NMAT);
  transpose_w_kernel<<<65536 / 256, 256, 0, stream>>>(r_W1, w1_t, 1); // rows 0..255

  // --- Router --------------------------------------------------------------
  router_gemm_kernel<<<gemm_blocks, 256, 131072, stream>>>(
      h_bf, w1_t, r_b1, r_W1, batch, glog, rh_bf, N_NODES);
  router_topk_kernel<<<(N_NODES + 255) / 256, 256, 0, stream>>>(
      rh_bf, r_W2, r_b2, sparse, N_NODES);

  // --- Layer-0 aggregation (shared across experts) -------------------------
  (void)hipMemsetAsync(A_f32, 0, NF * 4, stream);
  agg_kernel<<<agg_blocks, 256, 0, stream>>>(h_bf, src, dst, A_f32);
  f32_to_bf16_kernel<<<cvt_blocks, 256, 0, stream>>>(A_f32, agg0_bf, N_NODES * 32);

  // --- Per-expert chains (independent across experts) ----------------------
  for (int e = 0; e < N_EXP; ++e) {
    expert_gemm_kernel<<<gemm_blocks, 256, 262144, stream>>>(
        agg0_bf, h_bf,
        wrel_t  + (size_t)(0 * N_EXP + e) * 65536,
        wroot_t + (size_t)(0 * N_EXP + e) * 65536,
        brel + (size_t)(0 * N_EXP + e) * DH,
        Hc, N_NODES, /*relu=*/1);

    __bf16* cur = Hc; __bf16* nxt = Hn;
    for (int l = 1; l < L_LAYERS; ++l) {
      (void)hipMemsetAsync(A_f32, 0, NF * 4, stream);
      agg_kernel<<<agg_blocks, 256, 0, stream>>>(cur, src, dst, A_f32);
      f32_to_bf16_kernel<<<cvt_blocks, 256, 0, stream>>>(A_f32, A_bf, N_NODES * 32);
      expert_gemm_kernel<<<gemm_blocks, 256, 262144, stream>>>(
          A_bf, cur,
          wrel_t  + (size_t)(l * N_EXP + e) * 65536,
          wroot_t + (size_t)(l * N_EXP + e) * 65536,
          brel + (size_t)(l * N_EXP + e) * DH,
          nxt, N_NODES, /*relu=*/(l < L_LAYERS - 1) ? 1 : 0);
      __bf16* tmp = cur; cur = nxt; nxt = tmp;
    }

    combine_kernel<<<cvt_blocks, 256, 0, stream>>>(cur, sparse, e, out, N_NODES);
  }
}
